// KNNFeedForward_78340203479503
// MI455X (gfx1250) — compile-verified, matrix operand order checked by reading, not run on hardware
//
#include <hip/hip_runtime.h>
#include <hip/hip_bf16.h>
#include <cstdint>

typedef unsigned short u16;
typedef __attribute__((ext_vector_type(16))) __bf16 v16bf;
typedef __attribute__((ext_vector_type(8)))  float  v8f;

__device__ __forceinline__ u16 f2bf(float f) {
    unsigned int u = __float_as_uint(f);
    return (u16)((u + 0x7FFFu + ((u >> 16) & 1u)) >> 16);   // RNE
}

__device__ __forceinline__ v16bf load_frag(const u16* p) {
    // Per-lane 16 bf16 elements of one 16x32 (or 32x16^T) tile row:
    // elems 0..7 -> K = k0 + hi*8 + e ; elems 8..15 -> K = k0 + 16 + hi*8 + e
    union { v16bf v; uint4 u[2]; } f;
    f.u[0] = *(const uint4*)(p);
    f.u[1] = *(const uint4*)(p + 16);
    return f.v;
}

// ---------------------------------------------------------------------------
// fp32 -> bf16 bulk convert
// ---------------------------------------------------------------------------
__global__ void cvt_bf16(const float* __restrict__ s, u16* __restrict__ d, long long n) {
    long long i = (long long)blockIdx.x * blockDim.x + threadIdx.x;
    long long stride = (long long)gridDim.x * blockDim.x;
    for (; i < n; i += stride) d[i] = f2bf(s[i]);
}

// ---------------------------------------------------------------------------
// Tiny nets: pooled mean -> k_cont (B,3) and branch weights w (B,3)
// one block per batch
// ---------------------------------------------------------------------------
__global__ __launch_bounds__(256) void small_nets(
    const float* __restrict__ x,
    const float* __restrict__ k1w, const float* __restrict__ k1b,
    const float* __restrict__ k2w, const float* __restrict__ k2b,
    const float* __restrict__ w1w, const float* __restrict__ w1b,
    const float* __restrict__ w2w, const float* __restrict__ w2b,
    float* __restrict__ kc, float* __restrict__ wv)
{
    __shared__ float pooled[512];
    __shared__ float hid[256];
    __shared__ float tmp[8];
    const int b = blockIdx.x, t = threadIdx.x;

    for (int c = t; c < 512; c += 256) {
        float s = 0.f;
        const float* px = x + ((size_t)b * 1024) * 512 + c;
        for (int n = 0; n < 1024; ++n) s += px[(size_t)n * 512];
        pooled[c] = s * (1.0f / 1024.0f);
    }
    __syncthreads();

    if (t < 128) {
        float s = k1b[t];
        const float* wr = k1w + (size_t)t * 512;
        for (int c = 0; c < 512; ++c) s += pooled[c] * wr[c];
        hid[t] = fmaxf(s, 0.f);
    } else {
        int j = t - 128;
        float s = w1b[j];
        const float* wr = w1w + (size_t)j * 512;
        for (int c = 0; c < 512; ++c) s += pooled[c] * wr[c];
        hid[t] = fmaxf(s, 0.f);
    }
    __syncthreads();

    if (t < 3) {
        float s = k2b[t];
        const float* wr = k2w + (size_t)t * 128;
        for (int j = 0; j < 128; ++j) s += hid[j] * wr[j];
        float ratio = 1.f / (1.f + __expf(-s));
        kc[b * 3 + t] = 1.0f + ratio * 11.0f;           // K_MIN + r*(K_MAX-K_MIN)
    } else if (t >= 32 && t < 35) {
        int i = t - 32;
        float s = w2b[i];
        const float* wr = w2w + (size_t)i * 128;
        for (int j = 0; j < 128; ++j) s += hid[128 + j] * wr[j];
        tmp[i] = s;
    }
    __syncthreads();

    if (t == 0) {
        float mx = fmaxf(tmp[0], fmaxf(tmp[1], tmp[2]));
        float e0 = __expf(tmp[0] - mx), e1 = __expf(tmp[1] - mx), e2 = __expf(tmp[2] - mx);
        float inv = 1.f / (e0 + e1 + e2);
        wv[b * 3 + 0] = e0 * inv; wv[b * 3 + 1] = e1 * inv; wv[b * 3 + 2] = e2 * inv;
    }
}

// ---------------------------------------------------------------------------
// WMMA bf16 GEMM: C = A(MxK) * B(NxK)^T  (both row-major over K)
// Block = 8 waves, block tile 128x128, wave tile 32x64 (2x4 WMMA tiles).
// k-loop unrolled by 2 with a compile-time ping-pong buffer pair: loads for
// the next 32-K stage are issued before the current stage's WMMAs, and the
// alternating buffers remove all cross-iteration register copies.
// Requires K % 64 == 0 (holds: 512 / 2048 / 1024).
// MODE 0: bias+relu -> bf16 Cb          (fc1)
// MODE 1: bias -> bf16 Cb + bf16 Ct=C^T (fc2; Ct laid out (B,512,1024))
// MODE 2: -> fp32 Cf, batched by z      (sim)
// MODE 3: -> fp32 Cf, batched by z      (final aggregation -> d_out)
// ---------------------------------------------------------------------------
#define LOAD_STAGE(s, k)                                                      \
    do {                                                                      \
        ca0[s] = load_frag(ap0 + (k));                                        \
        ca1[s] = load_frag(ap1 + (k));                                        \
        _Pragma("unroll")                                                     \
        for (int j = 0; j < 4; ++j) cb[s][j] = load_frag(bp[j] + (k));        \
    } while (0)

#define WMMA_STAGE(s)                                                         \
    do {                                                                      \
        _Pragma("unroll")                                                     \
        for (int j = 0; j < 4; ++j) {                                         \
            acc[0][j] = __builtin_amdgcn_wmma_f32_16x16x32_bf16(              \
                false, ca0[s], false, cb[s][j], (short)0, acc[0][j],          \
                false, false);                                                \
            acc[1][j] = __builtin_amdgcn_wmma_f32_16x16x32_bf16(              \
                false, ca1[s], false, cb[s][j], (short)0, acc[1][j],          \
                false, false);                                                \
        }                                                                     \
    } while (0)

template <int MODE>
__global__ __launch_bounds__(256) void gemm_bf16(
    const u16* __restrict__ A, const u16* __restrict__ B,
    int M, int N, int K,
    long long sA, long long sB, long long sC,
    const float* __restrict__ bias,
    float* __restrict__ Cf, u16* __restrict__ Cb, u16* __restrict__ Ct)
{
    const int lane = threadIdx.x & 31;
    const int wid  = threadIdx.x >> 5;
    const int r  = lane & 15;
    const int hi = lane >> 4;

    A += (size_t)blockIdx.z * sA;
    B += (size_t)blockIdx.z * sB;

    const int m0 = blockIdx.y * 128 + (wid & 3) * 32;   // 4 wave-rows
    const int n0 = blockIdx.x * 128 + (wid >> 2) * 64;  // 2 wave-cols

    const u16* ap0 = A + (size_t)(m0 + r) * K + hi * 8;
    const u16* ap1 = A + (size_t)(m0 + 16 + r) * K + hi * 8;
    const u16* bp[4];
#pragma unroll
    for (int j = 0; j < 4; ++j) bp[j] = B + (size_t)(n0 + 16 * j + r) * K + hi * 8;

    v8f acc[2][4];
#pragma unroll
    for (int i = 0; i < 2; ++i)
#pragma unroll
        for (int j = 0; j < 4; ++j)
            acc[i][j] = (v8f){0.f, 0.f, 0.f, 0.f, 0.f, 0.f, 0.f, 0.f};

    v16bf ca0[2], ca1[2], cb[2][4];

    LOAD_STAGE(0, 0);
    int k0 = 0;
    for (; k0 + 64 < K; k0 += 64) {
        LOAD_STAGE(1, k0 + 32);   // in flight while stage-0 WMMAs run
        WMMA_STAGE(0);
        LOAD_STAGE(0, k0 + 64);   // in flight while stage-1 WMMAs run
        WMMA_STAGE(1);
    }
    // tail: k0 == K - 64
    LOAD_STAGE(1, k0 + 32);
    WMMA_STAGE(0);
    WMMA_STAGE(1);

    // Epilogue. C layout: vgpr e, lane -> (m = hi*8 + e, n = r) within 16x16 tile.
#pragma unroll
    for (int i = 0; i < 2; ++i) {
        const int mb = m0 + 16 * i + hi * 8;
#pragma unroll
        for (int j = 0; j < 4; ++j) {
            const int nn = n0 + 16 * j + r;
            float bv = 0.f;
            if (MODE <= 1) bv = bias[nn];
#pragma unroll
            for (int e = 0; e < 8; ++e) {
                const int mm = mb + e;
                float v = acc[i][j][e] + bv;
                if (MODE == 0) v = fmaxf(v, 0.f);
                if (MODE <= 1) Cb[(size_t)mm * N + nn] = f2bf(v);
                if (MODE == 1) {
                    const int bb = mm >> 10, ml = mm & 1023;
                    Ct[((size_t)bb * N + nn) * 1024 + ml] = f2bf(v);
                }
                if (MODE >= 2)
                    Cf[(size_t)blockIdx.z * sC + (size_t)mm * N + nn] = v;
            }
        }
    }
}

// ---------------------------------------------------------------------------
// Per-row: softmax + exact rank (stable ties) + 3 sigmoid gates blended by w,
// normalized per branch; emits blended attention row in bf16.
// One block per (b, n) row; 256 threads, 4 elements each.
// The 4KB sim row is pulled into LDS with the CDNA5 async global->LDS path
// (GLOBAL_LOAD_ASYNC_TO_LDS_B128, tracked by ASYNCcnt) — no VGPR staging.
// ---------------------------------------------------------------------------
__global__ __launch_bounds__(256) void row_attn(
    const float* __restrict__ S, const float* __restrict__ kc,
    const float* __restrict__ wv, u16* __restrict__ Ab)
{
    __shared__ float srow[1024];
    __shared__ float red[256];
    const int bi = blockIdx.x;
    const int b  = bi >> 10;
    const int t  = threadIdx.x;
    const float* Sr = S + (size_t)bi * 1024;

    {
        // async copy: each lane moves 16B; 256 lanes cover the 4KB row.
        // Generic shared pointer's low 32 bits are the LDS byte offset
        // (flat->LDS address truncation, ISA 10.2).
        unsigned lds_dst = (unsigned)(uintptr_t)(const void*)&srow[4 * t];
        unsigned long long gsrc = (unsigned long long)(uintptr_t)(Sr + 4 * t);
        asm volatile("global_load_async_to_lds_b128 %0, %1, off"
                     :: "v"(lds_dst), "v"(gsrc) : "memory");
        asm volatile("s_wait_asynccnt 0x0" ::: "memory");
    }
    __syncthreads();

    // row max
    float lm = -3.4e38f;
    for (int q = 0; q < 4; ++q) lm = fmaxf(lm, srow[t + 256 * q]);
    red[t] = lm; __syncthreads();
    for (int off = 128; off > 0; off >>= 1) {
        if (t < off) red[t] = fmaxf(red[t], red[t + off]);
        __syncthreads();
    }
    const float rowmax = red[0]; __syncthreads();

    // softmax numerators + sum
    float e[4]; float ls = 0.f;
    for (int q = 0; q < 4; ++q) { e[q] = __expf(srow[t + 256 * q] - rowmax); ls += e[q]; }
    red[t] = ls; __syncthreads();
    for (int off = 128; off > 0; off >>= 1) {
        if (t < off) red[t] += red[t + off];
        __syncthreads();
    }
    const float invp = 1.f / red[0]; __syncthreads();

    const float kcv[3] = {kc[b * 3 + 0], kc[b * 3 + 1], kc[b * 3 + 2]};
    float pm[4], g[4][3];
    float den[3] = {0.f, 0.f, 0.f};

    for (int q = 0; q < 4; ++q) {
        const int m = t + 256 * q;
        const float sm = srow[m];
        int cnt = 0;                              // rank = argsort(argsort(-sim))
        for (int j = 0; j < 1024; ++j) {
            const float sj = srow[j];
            cnt += (sj > sm) || ((sj == sm) && (j < m));
        }
        const float rf = (float)cnt + 0.5f;
        pm[q] = e[q] * invp;
        for (int i = 0; i < 3; ++i) {
            const float z = 12.0f * (kcv[i] - rf);    // STE_ALPHA = 12
            const float gg = 1.f / (1.f + __expf(-z));
            g[q][i] = gg;
            den[i] += pm[q] * gg;
        }
    }

    float invden[3];
    for (int i = 0; i < 3; ++i) {
        red[t] = den[i]; __syncthreads();
        for (int off = 128; off > 0; off >>= 1) {
            if (t < off) red[t] += red[t + off];
            __syncthreads();
        }
        invden[i] = 1.f / (red[0] + 1e-8f); __syncthreads();
    }

    const float wvv[3] = {wv[b * 3 + 0], wv[b * 3 + 1], wv[b * 3 + 2]};
    u16* Ar = Ab + (size_t)bi * 1024;
    for (int q = 0; q < 4; ++q) {
        float a = 0.f;
        for (int i = 0; i < 3; ++i) a += wvv[i] * pm[q] * g[q][i] * invden[i];
        Ar[t + 256 * q] = f2bf(a);
    }
}

// ---------------------------------------------------------------------------
extern "C" void kernel_launch(void* const* d_in, const int* in_sizes, int n_in,
                              void* d_out, int out_size, void* d_ws, size_t ws_size,
                              hipStream_t stream)
{
    (void)in_sizes; (void)n_in; (void)out_size; (void)ws_size;
    const float* x    = (const float*)d_in[0];
    const float* fc1w = (const float*)d_in[1];
    const float* fc1b = (const float*)d_in[2];
    const float* fc2w = (const float*)d_in[3];
    const float* fc2b = (const float*)d_in[4];
    const float* k1w  = (const float*)d_in[5];
    const float* k1b  = (const float*)d_in[6];
    const float* k2w  = (const float*)d_in[7];
    const float* k2b  = (const float*)d_in[8];
    const float* w1w  = (const float*)d_in[9];
    const float* w1b  = (const float*)d_in[10];
    const float* w2w  = (const float*)d_in[11];
    const float* w2b  = (const float*)d_in[12];

    char* ws = (char*)d_ws;
    size_t o = 0;
    auto alloc = [&](size_t bytes) { size_t c = o; o += (bytes + 255) & ~(size_t)255; return c; };
    u16*   XB  = (u16*)  (ws + alloc(4194304ull  * 2));   // x bf16       (8192,512)
    u16*   W1B = (u16*)  (ws + alloc(1048576ull  * 2));   // fc1_w bf16   (2048,512)
    u16*   W2B = (u16*)  (ws + alloc(1048576ull  * 2));   // fc2_w bf16   (512,2048)
    u16*   H1  = (u16*)  (ws + alloc(16777216ull * 2));   // relu hidden  (8192,2048)
    u16*   HB  = (u16*)  (ws + alloc(4194304ull  * 2));   // h bf16       (8192,512)
    u16*   HBT = (u16*)  (ws + alloc(4194304ull  * 2));   // h^T bf16     (8,512,1024)
    float* Sm  = (float*)(ws + alloc(8388608ull  * 4));   // sim fp32     (8,1024,1024)
    u16*   AB  = (u16*)  (ws + alloc(8388608ull  * 2));   // blended attn (8,1024,1024)
    float* KC  = (float*)(ws + alloc(256));               // k_cont (8,3)
    float* WV  = (float*)(ws + alloc(256));               // branch weights (8,3)

    cvt_bf16<<<4096, 256, 0, stream>>>(x,    XB,  4194304ll);
    cvt_bf16<<<1024, 256, 0, stream>>>(fc1w, W1B, 1048576ll);
    cvt_bf16<<<1024, 256, 0, stream>>>(fc2w, W2B, 1048576ll);

    small_nets<<<8, 256, 0, stream>>>(x, k1w, k1b, k2w, k2b, w1w, w1b, w2w, w2b, KC, WV);

    // h1 = relu(x @ fc1_w^T + b1)                 M=8192 N=2048 K=512
    gemm_bf16<0><<<dim3(16, 64, 1), 256, 0, stream>>>(
        XB, W1B, 8192, 2048, 512, 0, 0, 0, fc1b, nullptr, H1, nullptr);

    // h = h1 @ fc2_w^T + b2 (+ transposed copy)   M=8192 N=512 K=2048
    gemm_bf16<1><<<dim3(4, 64, 1), 256, 0, stream>>>(
        H1, W2B, 8192, 512, 2048, 0, 0, 0, fc2b, nullptr, HB, HBT);

    // sim = h @ h^T per batch                     M=N=1024 K=512, z=8
    gemm_bf16<2><<<dim3(8, 8, 8), 256, 0, stream>>>(
        HB, HB, 1024, 1024, 512,
        1024ll * 512, 1024ll * 512, 1024ll * 1024,
        nullptr, Sm, nullptr, nullptr);

    // blended, normalized gated attention rows
    row_attn<<<8192, 256, 0, stream>>>(Sm, KC, WV, AB);

    // y = attn_blend @ h per batch                M=1024 N=512 K=1024, z=8
    gemm_bf16<3><<<dim3(4, 8, 8), 256, 0, stream>>>(
        AB, HBT, 1024, 512, 1024,
        1024ll * 1024, 512ll * 1024, 1024ll * 512,
        nullptr, (float*)d_out, nullptr, nullptr);
}
